// CausalSelfAttention_84473416777946
// MI455X (gfx1250) — compile-verified, matrix operand order checked by reading, not run on hardware
//
#include <hip/hip_runtime.h>

typedef __attribute__((ext_vector_type(16))) _Float16 v16h;
typedef __attribute__((ext_vector_type(8)))  float    v8f;

#define WMMA_F16(A, B, C) \
    __builtin_amdgcn_wmma_f32_16x16x32_f16(false, (A), false, (B), (short)0, (C), false, false)

// ---------------------------------------------------------------------------
// CDNA5 16-bit A-frag (16x32, MxK) per-lane layout:
//   lane L: row M = L&15, khalf = L>>4
//   element e (vgpr v=e>>1, pos=e&1): K = (v<4?0:16) + khalf*8 + (v&3)*2 + pos
// => per lane: K runs khalf*8+0..7 and khalf*8+16..23, i.e. two 32-byte
//    contiguous f32 runs -> four global_load_b128 per fragment.
// B-frag (32x16, KxN) mirrors this with N = L&15 per lane.
// ---------------------------------------------------------------------------
static __device__ __forceinline__
v16h load_frag_rowk(const float* __restrict__ base, long ld, int idx0, int k0,
                    int lane, float scale) {
    int idx   = idx0 + (lane & 15);
    int khalf = lane >> 4;
    const float* p = base + (long)idx * ld + k0 + khalf * 8;
    float4 x0 = *(const float4*)(p);
    float4 x1 = *(const float4*)(p + 4);
    float4 x2 = *(const float4*)(p + 16);
    float4 x3 = *(const float4*)(p + 20);
    v16h f;
    f[0]  = (_Float16)(x0.x * scale); f[1]  = (_Float16)(x0.y * scale);
    f[2]  = (_Float16)(x0.z * scale); f[3]  = (_Float16)(x0.w * scale);
    f[4]  = (_Float16)(x1.x * scale); f[5]  = (_Float16)(x1.y * scale);
    f[6]  = (_Float16)(x1.z * scale); f[7]  = (_Float16)(x1.w * scale);
    f[8]  = (_Float16)(x2.x * scale); f[9]  = (_Float16)(x2.y * scale);
    f[10] = (_Float16)(x2.z * scale); f[11] = (_Float16)(x2.w * scale);
    f[12] = (_Float16)(x3.x * scale); f[13] = (_Float16)(x3.y * scale);
    f[14] = (_Float16)(x3.z * scale); f[15] = (_Float16)(x3.w * scale);
    return f;
}

// Strided along K (V as B-frag: column n fixed per lane, K walks rows).
static __device__ __forceinline__
v16h load_bfrag_strided(const float* __restrict__ base, long ld, int n0, int k0,
                        int lane) {
    int n     = n0 + (lane & 15);
    int khalf = lane >> 4;
    const float* p = base + n;
    v16h f;
#pragma unroll
    for (int v = 0; v < 8; ++v) {
        int kb = ((v < 4) ? 0 : 16) + khalf * 8 + (v & 3) * 2;
        f[2 * v]     = (_Float16)p[(long)(k0 + kb)     * ld];
        f[2 * v + 1] = (_Float16)p[(long)(k0 + kb + 1) * ld];
    }
    return f;
}

// ---------------------------------------------------------------------------
// out[M,N] = A[M,K] @ W[N,K]^T + bias[N]   (torch Linear)
// Register-blocked: each wave computes a 32(M) x 64(N) tile with 8
// accumulators; per 32-wide k-step: 2 A-frags + 4 B-frags -> 8 WMMAs
// (~22 FLOP/byte of fragment traffic). 4 waves/block cover 128x64.
// ---------------------------------------------------------------------------
__global__ void gemm_bias_wmma(const float* __restrict__ A,
                               const float* __restrict__ W,
                               const float* __restrict__ bias,
                               float* __restrict__ out,
                               int M, int N, int K) {
    int lane  = threadIdx.x & 31;
    int wave  = threadIdx.x >> 5;
    int khalf = lane >> 4;
    int m0 = blockIdx.y * 128 + wave * 32;
    int n0 = blockIdx.x * 64;

    v8f c[2][4] = {};
    for (int k0 = 0; k0 < K; k0 += 32) {
        v16h a0 = load_frag_rowk(A, K, m0,      k0, lane, 1.0f);
        v16h a1 = load_frag_rowk(A, K, m0 + 16, k0, lane, 1.0f);
        v16h b0 = load_frag_rowk(W, K, n0,      k0, lane, 1.0f);
        v16h b1 = load_frag_rowk(W, K, n0 + 16, k0, lane, 1.0f);
        v16h b2 = load_frag_rowk(W, K, n0 + 32, k0, lane, 1.0f);
        v16h b3 = load_frag_rowk(W, K, n0 + 48, k0, lane, 1.0f);
        c[0][0] = WMMA_F16(a0, b0, c[0][0]);
        c[0][1] = WMMA_F16(a0, b1, c[0][1]);
        c[0][2] = WMMA_F16(a0, b2, c[0][2]);
        c[0][3] = WMMA_F16(a0, b3, c[0][3]);
        c[1][0] = WMMA_F16(a1, b0, c[1][0]);
        c[1][1] = WMMA_F16(a1, b1, c[1][1]);
        c[1][2] = WMMA_F16(a1, b2, c[1][2]);
        c[1][3] = WMMA_F16(a1, b3, c[1][3]);
    }
    // C/D layout: lane holds col n=L&15, rows M=(L>>4)*8 + r
#pragma unroll
    for (int nj = 0; nj < 4; ++nj) {
        int n  = n0 + nj * 16 + (lane & 15);
        float bs = bias[n];
#pragma unroll
        for (int mi = 0; mi < 2; ++mi) {
            int mr = m0 + mi * 16 + khalf * 8;
#pragma unroll
            for (int r = 0; r < 8; ++r)
                out[(long)(mr + r) * N + n] = c[mi][nj][r] + bs;
        }
    }
}

// ---------------------------------------------------------------------------
// Fused causal flash attention. One wave per (b, h, 32-query strip = 2 tiles).
// Computes S^T = K Q^T so softmax reductions are per-lane + one shfl_xor(16),
// and the S^T accumulator layout maps element-for-element onto the A-fragment
// of P for the P*V WMMA (no LDS / no shuffles between the two GEMMs).
// K/V fragments are loaded once per chunk and reused by both query tiles.
// ---------------------------------------------------------------------------
__global__ void attn_flash_wmma(const float* __restrict__ qkv,
                                float* __restrict__ y,
                                int T, int C, int H) {
    const int S3 = 3 * C;        // qkv row stride
    const int Dh = C / H;        // 64
    int lane  = threadIdx.x & 31;
    int khalf = lane >> 4;
    int qt0 = blockIdx.x * 2;    // first of two query tiles
    int qt1 = qt0 + 1;
    int bh  = blockIdx.y;
    int b   = bh / H, h = bh % H;

    long base = (long)b * T * S3;
    const float* Qp = qkv + base +         h * Dh;   // Q[t][d] at Qp[t*S3+d]
    const float* Kp = qkv + base + C     + h * Dh;
    const float* Vp = qkv + base + 2 * C + h * Dh;

    float scale = rsqrtf((float)Dh);
    int qq[2]  = { qt0 * 16 + (lane & 15), qt1 * 16 + (lane & 15) };

    // Q^T as B-fragments (scale folded in), per qtile x per 32-wide d-chunk
    v16h bq[2][2];
    bq[0][0] = load_frag_rowk(Qp, S3, qt0 * 16, 0,  lane, scale);
    bq[0][1] = load_frag_rowk(Qp, S3, qt0 * 16, 32, lane, scale);
    bq[1][0] = load_frag_rowk(Qp, S3, qt1 * 16, 0,  lane, scale);
    bq[1][1] = load_frag_rowk(Qp, S3, qt1 * 16, 32, lane, scale);

    float m[2] = { -1e30f, -1e30f };
    float l[2] = { 0.0f, 0.0f };
    v8f o[2][4] = {};

    // qt1 is odd, so ct = 0,2,...,qt1-1 covers key tiles 0..qt1 exactly.
    for (int ct = 0; ct <= qt1; ct += 2) {
        // ---- K fragments for tiles ct, ct+1 (shared by both qtiles) --------
        v16h ak00 = load_frag_rowk(Kp, S3,  ct      * 16, 0,  lane, 1.0f);
        v16h ak01 = load_frag_rowk(Kp, S3,  ct      * 16, 32, lane, 1.0f);
        v16h ak10 = load_frag_rowk(Kp, S3, (ct + 1) * 16, 0,  lane, 1.0f);
        v16h ak11 = load_frag_rowk(Kp, S3, (ct + 1) * 16, 32, lane, 1.0f);

        v16h ap[2];
#pragma unroll
        for (int j = 0; j < 2; ++j) {
            // ---- S^T tiles: rows = keys, cols = queries --------------------
            v8f s0v = {}, s1v = {};
            s0v = WMMA_F16(ak00, bq[j][0], s0v);
            s0v = WMMA_F16(ak01, bq[j][1], s0v);
            s1v = WMMA_F16(ak10, bq[j][0], s1v);
            s1v = WMMA_F16(ak11, bq[j][1], s1v);

            // ---- causal mask + running max ---------------------------------
            float s0[8], s1[8];
            float lm = -1e30f;
#pragma unroll
            for (int r = 0; r < 8; ++r) {
                int key0 = ct * 16 + khalf * 8 + r;
                s0[r] = (key0      <= qq[j]) ? s0v[r] : -1e30f;
                s1[r] = (key0 + 16 <= qq[j]) ? s1v[r] : -1e30f;
                lm = fmaxf(lm, fmaxf(s0[r], s1[r]));
            }
            lm = fmaxf(lm, __shfl_xor(lm, 16, 32));   // partner lane, same q
            float mn    = fmaxf(m[j], lm);
            float alpha = __expf(m[j] - mn);

            // ---- p = exp(s - mn), packed straight into the P A-fragment ----
            float ls = 0.0f;
#pragma unroll
            for (int r = 0; r < 8; ++r) {
                float p0 = __expf(s0[r] - mn);
                float p1 = __expf(s1[r] - mn);
                ls += p0 + p1;
                ap[j][r]     = (_Float16)p0;  // keys ct*16    + khalf*8 + r
                ap[j][8 + r] = (_Float16)p1;  // keys ct*16+16 + khalf*8 + r
            }
            ls += __shfl_xor(ls, 16, 32);
            l[j] = l[j] * alpha + ls;
            m[j] = mn;

            // ---- rescale O rows (O layout: lane = d-col, rows = queries) ---
            float arow[8];
#pragma unroll
            for (int r = 0; r < 8; ++r)
                arow[r] = __shfl(alpha, khalf * 8 + r, 32);
#pragma unroll
            for (int r = 0; r < 8; ++r) {
                o[j][0][r] *= arow[r]; o[j][1][r] *= arow[r];
                o[j][2][r] *= arow[r]; o[j][3][r] *= arow[r];
            }
        }

        // ---- O += P * V  (V fragments shared by both qtiles) ---------------
#pragma unroll
        for (int dt = 0; dt < 4; ++dt) {
            v16h bv = load_bfrag_strided(Vp, S3, dt * 16, ct * 16, lane);
            o[0][dt] = WMMA_F16(ap[0], bv, o[0][dt]);
            o[1][dt] = WMMA_F16(ap[1], bv, o[1][dt]);
        }
    }

    // ---- normalize by l and store y[b, q, h*Dh + d] ------------------------
    int n = lane & 15;
#pragma unroll
    for (int j = 0; j < 2; ++j) {
        float lrow[8];
#pragma unroll
        for (int r = 0; r < 8; ++r)
            lrow[r] = __shfl(l[j], khalf * 8 + r, 32);
#pragma unroll
        for (int r = 0; r < 8; ++r) {
            long row = (long)(b * T + (qt0 + j) * 16 + khalf * 8 + r) * C + h * Dh;
            float inv = 1.0f / lrow[r];
            y[row +      n] = o[j][0][r] * inv;
            y[row + 16 + n] = o[j][1][r] * inv;
            y[row + 32 + n] = o[j][2][r] * inv;
            y[row + 48 + n] = o[j][3][r] * inv;
        }
    }
}

// ---------------------------------------------------------------------------
extern "C" void kernel_launch(void* const* d_in, const int* in_sizes, int n_in,
                              void* d_out, int out_size, void* d_ws, size_t ws_size,
                              hipStream_t stream) {
    const float* x      = (const float*)d_in[0];
    const float* qkv_w  = (const float*)d_in[1];
    const float* qkv_b  = (const float*)d_in[2];
    const float* proj_w = (const float*)d_in[3];
    const float* proj_b = (const float*)d_in[4];
    float* out = (float*)d_out;

    const int Bsz = 4, T = 2048, C = 1024, H = 16;
    const int M  = Bsz * T;      // 8192 rows
    const int N1 = 3 * C;        // 3072

    float* qkv = (float*)d_ws;                 // [M, 3C]  96 MB
    float* y   = qkv + (size_t)M * N1;         // [M, C]   32 MB

    // 1) QKV projection: grid covers 128-row x 64-col block tiles
    gemm_bias_wmma<<<dim3(N1 / 64, M / 128), 128, 0, stream>>>(
        x, qkv_w, qkv_b, qkv, M, N1, C);

    // 2) fused causal attention (flash-style, 32 queries per wave)
    attn_flash_wmma<<<dim3(T / 32, Bsz * H), 32, 0, stream>>>(qkv, y, T, C, H);

    // 3) output projection
    gemm_bias_wmma<<<dim3(C / 64, M / 128), 128, 0, stream>>>(
        y, proj_w, proj_b, out, M, C, C);
}
